// NestedTransformerEncoderLayer_42451456753846
// MI455X (gfx1250) — matrix operand order repaired and checked by reading, not compiled, not run on hardware
//
#include <hip/hip_runtime.h>
#include <hip/hip_bf16.h>

// ---------------------------------------------------------------------------
// NestedTransformerEncoderLayer for MI455X (gfx1250), bf16 WMMA GEMMs.
// D=256, H=8, HD=32, NB=8, DFF=1024, Npix=131072
// ---------------------------------------------------------------------------

typedef __attribute__((ext_vector_type(16))) __bf16        v16bf;
typedef __attribute__((ext_vector_type(8)))  float         v8f;
typedef __attribute__((ext_vector_type(8)))  unsigned short us8;
typedef __attribute__((ext_vector_type(16))) unsigned short us16;

static __device__ __forceinline__ unsigned short f2bf(float f) {
    unsigned int u = __float_as_uint(f);
    u += 0x7FFFu + ((u >> 16) & 1u);           // round-to-nearest-even
    return (unsigned short)(u >> 16);
}
static __device__ __forceinline__ float bf2f(unsigned short h) {
    return __uint_as_float(((unsigned int)h) << 16);
}

// ---------------------------------------------------------------------------
// Weight prep: cast fp32 KxN row-major -> bf16 NxK (column-major B) so that a
// WMMA B fragment is a single contiguous 32B load per lane.
// ---------------------------------------------------------------------------
__global__ void wprep_kernel(const float* __restrict__ W, unsigned short* __restrict__ Wt,
                             int K, int N) {
    size_t idx = (size_t)blockIdx.x * 256 + threadIdx.x;
    if (idx >= (size_t)K * N) return;
    int n = (int)(idx / K), k = (int)(idx % K);
    Wt[idx] = f2bf(W[(size_t)k * N + n]);
}

__global__ void wcat_prep_kernel(const float* __restrict__ wk, const float* __restrict__ wqkv,
                                 unsigned short* __restrict__ Wt) {
    size_t idx = (size_t)blockIdx.x * 256 + threadIdx.x;  // 1024*256
    int n = (int)(idx / 256), k = (int)(idx % 256);
    float v = (n < 256) ? wk[(size_t)k * 256 + n] : wqkv[(size_t)k * 768 + (n - 256)];
    Wt[idx] = f2bf(v);
}

__global__ void bcat_prep_kernel(const float* __restrict__ bk, const float* __restrict__ bqkv,
                                 float* __restrict__ bc) {
    int i = blockIdx.x * 256 + threadIdx.x;
    if (i < 1024) bc[i] = (i < 256) ? bk[i] : bqkv[i - 256];
}

// ---------------------------------------------------------------------------
// Embed: xe = x @ w0 + b0, write fp32 (for residual) and bf16 (for GEMM A).
// ---------------------------------------------------------------------------
__global__ __launch_bounds__(256) void embed_kernel(
    const float* __restrict__ x, const float* __restrict__ w0, const float* __restrict__ b0,
    float* __restrict__ xef, unsigned short* __restrict__ xeb) {
    size_t n = blockIdx.x;
    int t = threadIdx.x;                       // channel d
    float x0 = x[n * 3 + 0], x1 = x[n * 3 + 1], x2 = x[n * 3 + 2];
    float v = x0 * w0[t] + x1 * w0[256 + t] + x2 * w0[512 + t] + b0[t];
    xef[n * 256 + t] = v;
    xeb[n * 256 + t] = f2bf(v);
}

// ---------------------------------------------------------------------------
// bf16 WMMA GEMM: C[M,N] = A[M,K](bf16,row) * Bt[N,K](bf16,"col-major B") + bias
// Workgroup 128x128 (8 waves, wave tile 32x64), K step 32, double-buffered.
// ---------------------------------------------------------------------------
struct Frags { v16bf a[2]; v16bf b[4]; };

static __device__ __forceinline__ void load_frags(
    Frags& f, const unsigned short* __restrict__ A, const unsigned short* __restrict__ Bt,
    int K, int m_base, int n_base, int lm16, int half, int k0) {
    // A fragment: contiguous K per lane half (half*16 .. +15), matching B
#pragma unroll
    for (int mt = 0; mt < 2; ++mt) {
        const unsigned short* p = A + (size_t)(m_base + mt * 16 + lm16) * K + k0 + half * 16;
        us8 lo = *(const us8*)p;            // elems 0..7  (K = base+0..7)
        us8 hi = *(const us8*)(p + 8);      // elems 8..15 (K = base+8..15)
        union { us16 u; v16bf v; } xu;
#pragma unroll
        for (int i = 0; i < 8; ++i) { xu.u[i] = lo[i]; xu.u[i + 8] = hi[i]; }
        f.a[mt] = xu.v;
    }
    // B fragment: lane holds column N=n_base+lm16, contiguous K (half*16 .. +15)
#pragma unroll
    for (int nt = 0; nt < 4; ++nt) {
        const unsigned short* p = Bt + (size_t)(n_base + nt * 16 + lm16) * K + k0 + half * 16;
        union { us16 u; v16bf v; } xu;
        xu.u = *(const us16*)p;
        f.b[nt] = xu.v;
    }
}

static __device__ __forceinline__ void mma_frags(v8f (&acc)[2][4], const Frags& f) {
#pragma unroll
    for (int mt = 0; mt < 2; ++mt)
#pragma unroll
        for (int nt = 0; nt < 4; ++nt)
            acc[mt][nt] = __builtin_amdgcn_wmma_f32_16x16x32_bf16(
                false, f.a[mt], false, f.b[nt], (short)0, acc[mt][nt], false, false);
}

template <bool RELU, bool OUTF, bool OUTB>
__global__ __launch_bounds__(256) void gemm_bf16_kernel(
    const unsigned short* __restrict__ A, const unsigned short* __restrict__ Bt,
    const float* __restrict__ bias, float* __restrict__ Cf,
    unsigned short* __restrict__ Cb, int M, int N, int K) {
    int tid  = threadIdx.x;
    int lane = tid & 31, wave = tid >> 5;
    int wm = wave & 3, wn = wave >> 2;          // 4 (M) x 2 (N) waves
    int lm16 = lane & 15, half = lane >> 4;
    int m_base = blockIdx.y * 128 + wm * 32;
    int n_base = blockIdx.x * 128 + wn * 64;

    v8f acc[2][4];
#pragma unroll
    for (int i = 0; i < 2; ++i)
#pragma unroll
        for (int j = 0; j < 4; ++j)
#pragma unroll
            for (int e = 0; e < 8; ++e) acc[i][j][e] = 0.0f;

    // K is always a multiple of 64 here (256 or 1024): double-buffered pipeline.
    Frags f0, f1;
    load_frags(f0, A, Bt, K, m_base, n_base, lm16, half, 0);
    int k0 = 0;
    for (; k0 < K - 64; k0 += 64) {
        load_frags(f1, A, Bt, K, m_base, n_base, lm16, half, k0 + 32);
        mma_frags(acc, f0);
        load_frags(f0, A, Bt, K, m_base, n_base, lm16, half, k0 + 64);
        mma_frags(acc, f1);
    }
    load_frags(f1, A, Bt, K, m_base, n_base, lm16, half, K - 32);
    mma_frags(acc, f0);
    mma_frags(acc, f1);

#pragma unroll
    for (int mt = 0; mt < 2; ++mt) {
#pragma unroll
        for (int nt = 0; nt < 4; ++nt) {
            int col = n_base + nt * 16 + lm16;
            float bv = bias[col];
#pragma unroll
            for (int j = 0; j < 8; ++j) {
                int rowi = m_base + mt * 16 + half * 8 + j;   // C layout: M=j+8*half
                float v = acc[mt][nt][j] + bv;
                if (RELU) v = fmaxf(v, 0.0f);
                size_t idx = (size_t)rowi * N + col;
                if (OUTF) Cf[idx] = v;
                if (OUTB) Cb[idx] = f2bf(v);
            }
        }
    }
}

// ---------------------------------------------------------------------------
// Attention stage 1: per (s,h,b): logits(2x256), a_row softmax(256),
// a_col 2-way softmax per l, lm[96] = a_row . qkvb, add pos -> q1/k1/v1.
// ---------------------------------------------------------------------------
__global__ __launch_bounds__(256) void attn_stage1_kernel(
    const unsigned short* __restrict__ catb, const float* __restrict__ q_embed,
    const float* __restrict__ pos_q1, const float* __restrict__ pos_k1,
    float* __restrict__ a_col, float* __restrict__ q1o,
    float* __restrict__ k1o, float* __restrict__ v1o) {
    int b = blockIdx.x, h = blockIdx.y, s = blockIdx.z;
    int t = threadIdx.x;                        // l in [0,256)
    __shared__ float ars[256];
    __shared__ float red[256];
    __shared__ unsigned short qkvs[256 * 96];   // 48KB

    int r = t >> 4, c = t & 15;
    int ii = (b >> 3) * 16 + r, jj = (b & 7) * 16 + c;
    size_t n = ((size_t)s * 128 + ii) * 128 + jj;
    const unsigned short* row = catb + n * 1024 + (size_t)h * 128;

    const float* qe = q_embed + ((size_t)(h * 64 + b) * 2) * 32;
    float d0 = 0.f, d1 = 0.f;
#pragma unroll
    for (int d = 0; d < 32; ++d) {
        float kv = bf2f(row[d]);
        d0 += qe[d] * kv;
        d1 += qe[32 + d] * kv;
    }
#pragma unroll
    for (int e = 0; e < 96; ++e) qkvs[t * 96 + e] = row[32 + e];

    // softmax over l of logits[q=0]
    red[t] = d0; __syncthreads();
    for (int st = 128; st > 0; st >>= 1) { if (t < st) red[t] = fmaxf(red[t], red[t + st]); __syncthreads(); }
    float mx = red[0]; __syncthreads();
    float e0 = __expf(d0 - mx);
    red[t] = e0; __syncthreads();
    for (int st = 128; st > 0; st >>= 1) { if (t < st) red[t] += red[t + st]; __syncthreads(); }
    float ar = e0 / red[0];
    ars[t] = ar;

    // a_col: 2-way softmax over q per l
    float mc = fmaxf(d0, d1);
    float c0 = __expf(d0 - mc), c1 = __expf(d1 - mc);
    float cs = 1.0f / (c0 + c1);
    size_t ab = (((size_t)(s * 8 + h) * 64 + b) * 256 + t) * 2;
    a_col[ab] = c0 * cs;
    a_col[ab + 1] = c1 * cs;
    __syncthreads();

    // lm[e] = sum_l a_row[l] * qkvb[l][e]
    if (t < 96) {
        float acc = 0.f;
        for (int l = 0; l < 256; ++l) acc += ars[l] * bf2f(qkvs[l * 96 + t]);
        size_t ob = ((size_t)(s * 8 + h) * 64 + b) * 32;
        int pb = (h * 64 + b) * 32;
        if (t < 32)      q1o[ob + t]        = acc + pos_q1[pb + t];
        else if (t < 64) k1o[ob + t - 32]   = acc + pos_k1[pb + t - 32];
        else             v1o[ob + t - 64]   = acc;
    }
}

// ---------------------------------------------------------------------------
// Attention stage 2: per (s,h): att = softmax(q1 k1^T / sqrt(32)) (64x64),
// o = att @ v1 (64x32).
// ---------------------------------------------------------------------------
__global__ __launch_bounds__(256) void attn_stage2_kernel(
    const float* __restrict__ q1o, const float* __restrict__ k1o,
    const float* __restrict__ v1o, float* __restrict__ obuf) {
    int h = blockIdx.x, s = blockIdx.y;
    int t = threadIdx.x;
    __shared__ float q1s[2048], k1s[2048], v1s[2048], att[4096];
    size_t base = (size_t)(s * 8 + h) * 2048;
    for (int i = t; i < 2048; i += 256) {
        q1s[i] = q1o[base + i]; k1s[i] = k1o[base + i]; v1s[i] = v1o[base + i];
    }
    __syncthreads();
    const float scale = 0.17677669529663687f;   // 1/sqrt(32)
    for (int idx = t; idx < 4096; idx += 256) {
        int i = idx >> 6, j = idx & 63;
        float acc = 0.f;
#pragma unroll
        for (int d = 0; d < 32; ++d) acc += q1s[i * 32 + d] * k1s[j * 32 + d];
        att[idx] = acc * scale;
    }
    __syncthreads();
    if (t < 64) {
        float mx = -1e30f;
        for (int j = 0; j < 64; ++j) mx = fmaxf(mx, att[t * 64 + j]);
        float sum = 0.f;
        for (int j = 0; j < 64; ++j) { float e = __expf(att[t * 64 + j] - mx); att[t * 64 + j] = e; sum += e; }
        float inv = 1.f / sum;
        for (int j = 0; j < 64; ++j) att[t * 64 + j] *= inv;
    }
    __syncthreads();
    for (int idx = t; idx < 2048; idx += 256) {
        int i = idx >> 5, d = idx & 31;
        float acc = 0.f;
#pragma unroll
        for (int j = 0; j < 64; ++j) acc += att[i * 64 + j] * v1s[j * 32 + d];
        obuf[base + idx] = acc;
    }
}

// ---------------------------------------------------------------------------
// tok writer: tok[pixel, h*32+d] = ac0*o[b,d] + ac1*o_z[h,b,d]  (bf16)
// with the inverse block permutation back to pixel order.
// ---------------------------------------------------------------------------
__global__ __launch_bounds__(256) void tok_write_kernel(
    const float* __restrict__ a_col, const float* __restrict__ obuf,
    const float* __restrict__ o_z, unsigned short* __restrict__ tokb) {
    int b = blockIdx.x, h = blockIdx.y, s = blockIdx.z;
    int t = threadIdx.x;                        // l
    __shared__ float os[32], ozs[32];
    size_t ob = ((size_t)(s * 8 + h) * 64 + b) * 32;
    int zb = (h * 64 + b) * 32;
    if (t < 32) { os[t] = obuf[ob + t]; ozs[t] = o_z[zb + t]; }
    __syncthreads();
    size_t ab = (((size_t)(s * 8 + h) * 64 + b) * 256 + t) * 2;
    float ac0 = a_col[ab], ac1 = a_col[ab + 1];
    int r = t >> 4, c = t & 15;
    int ii = (b >> 3) * 16 + r, jj = (b & 7) * 16 + c;
    size_t n = ((size_t)s * 128 + ii) * 128 + jj;
    unsigned short* dst = tokb + n * 256 + h * 32;
#pragma unroll
    for (int d = 0; d < 32; ++d) dst[d] = f2bf(ac0 * os[d] + ac1 * ozs[d]);
}

// ---------------------------------------------------------------------------
// LayerNorm over D=256: y = LN(a+b)*g+be ; optional bf16 copy.
// ---------------------------------------------------------------------------
template <bool OUTB>
__global__ __launch_bounds__(256) void ln_kernel(
    const float* __restrict__ a, const float* __restrict__ b,
    const float* __restrict__ g, const float* __restrict__ be,
    float* __restrict__ yf, unsigned short* __restrict__ yb) {
    size_t row = blockIdx.x;
    int t = threadIdx.x;
    __shared__ float red[256];
    float v = a[row * 256 + t] + b[row * 256 + t];
    red[t] = v; __syncthreads();
    for (int st = 128; st > 0; st >>= 1) { if (t < st) red[t] += red[t + st]; __syncthreads(); }
    float m = red[0] * (1.0f / 256.0f); __syncthreads();
    float dv = v - m;
    red[t] = dv * dv; __syncthreads();
    for (int st = 128; st > 0; st >>= 1) { if (t < st) red[t] += red[t + st]; __syncthreads(); }
    float var = red[0] * (1.0f / 256.0f);
    float y = dv * rsqrtf(var + 1e-5f) * g[t] + be[t];
    yf[row * 256 + t] = y;
    if (OUTB) yb[row * 256 + t] = f2bf(y);
}

// ---------------------------------------------------------------------------
// Host side
// ---------------------------------------------------------------------------
extern "C" void kernel_launch(void* const* d_in, const int* in_sizes, int n_in,
                              void* d_out, int out_size, void* d_ws, size_t ws_size,
                              hipStream_t stream) {
    (void)in_sizes; (void)n_in; (void)out_size; (void)ws_size;
    const float* x      = (const float*)d_in[0];
    const float* w0     = (const float*)d_in[1];
    const float* b0     = (const float*)d_in[2];
    const float* wk     = (const float*)d_in[3];
    const float* bk     = (const float*)d_in[4];
    const float* wqkv   = (const float*)d_in[5];
    const float* bqkv   = (const float*)d_in[6];
    const float* qemb   = (const float*)d_in[7];
    const float* o_z    = (const float*)d_in[8];
    const float* pos_q1 = (const float*)d_in[9];
    const float* pos_k1 = (const float*)d_in[10];
    const float* wo     = (const float*)d_in[11];
    const float* bo     = (const float*)d_in[12];
    const float* g1     = (const float*)d_in[13];
    const float* be1    = (const float*)d_in[14];
    const float* w1     = (const float*)d_in[15];
    const float* b1     = (const float*)d_in[16];
    const float* w2     = (const float*)d_in[17];
    const float* b2     = (const float*)d_in[18];
    const float* g2     = (const float*)d_in[19];
    const float* be2    = (const float*)d_in[20];

    constexpr size_t NPIX = 8ull * 128 * 128;   // 131072

    char* ws = (char*)d_ws;
    size_t off = 0;
    auto alloc = [&](size_t bytes) { size_t r = off; off = (off + bytes + 255) & ~(size_t)255; return r; };

    float*          xef   = (float*)(ws + alloc(NPIX * 256 * 4));
    unsigned short* xeb   = (unsigned short*)(ws + alloc(NPIX * 256 * 2));   // reused for y_bf16
    unsigned short* catb  = (unsigned short*)(ws + alloc(NPIX * 1024 * 2));  // reused for relu hidden
    unsigned short* tokb  = (unsigned short*)(ws + alloc(NPIX * 256 * 2));
    float*          resf  = (float*)(ws + alloc(NPIX * 256 * 4));            // reused for MLP out
    float*          yf    = (float*)(ws + alloc(NPIX * 256 * 4));
    unsigned short* wcatt = (unsigned short*)(ws + alloc(1024 * 256 * 2));
    float*          bcat  = (float*)(ws + alloc(1024 * 4));
    unsigned short* wot   = (unsigned short*)(ws + alloc(256 * 256 * 2));
    unsigned short* w1t   = (unsigned short*)(ws + alloc(1024 * 256 * 2));
    unsigned short* w2t   = (unsigned short*)(ws + alloc(256 * 1024 * 2));
    float*          acol  = (float*)(ws + alloc(8ull * 8 * 64 * 256 * 2 * 4));
    float*          q1o   = (float*)(ws + alloc(8ull * 8 * 64 * 32 * 4));
    float*          k1o   = (float*)(ws + alloc(8ull * 8 * 64 * 32 * 4));
    float*          v1o   = (float*)(ws + alloc(8ull * 8 * 64 * 32 * 4));
    float*          obuf  = (float*)(ws + alloc(8ull * 8 * 64 * 32 * 4));

    // --- weight prep ---
    wcat_prep_kernel<<<1024, 256, 0, stream>>>(wk, wqkv, wcatt);
    bcat_prep_kernel<<<4, 256, 0, stream>>>(bk, bqkv, bcat);
    wprep_kernel<<<256, 256, 0, stream>>>(wo, wot, 256, 256);
    wprep_kernel<<<1024, 256, 0, stream>>>(w1, w1t, 256, 1024);
    wprep_kernel<<<1024, 256, 0, stream>>>(w2, w2t, 1024, 256);

    // --- embed ---
    embed_kernel<<<(unsigned)NPIX, 256, 0, stream>>>(x, w0, b0, xef, xeb);

    // --- cat = xe @ [wk|wqkv] + [bk|bqkv]  (M=131072, N=1024, K=256) -> bf16 ---
    gemm_bf16_kernel<false, false, true><<<dim3(1024 / 128, NPIX / 128), 256, 0, stream>>>(
        xeb, wcatt, bcat, nullptr, catb, (int)NPIX, 1024, 256);

    // --- block attention ---
    attn_stage1_kernel<<<dim3(64, 8, 8), 256, 0, stream>>>(
        catb, qemb, pos_q1, pos_k1, acol, q1o, k1o, v1o);
    attn_stage2_kernel<<<dim3(8, 8), 256, 0, stream>>>(q1o, k1o, v1o, obuf);
    tok_write_kernel<<<dim3(64, 8, 8), 256, 0, stream>>>(acol, obuf, o_z, tokb);

    // --- res = tok @ wo + bo  (N=256, K=256) -> f32 ---
    gemm_bf16_kernel<false, true, false><<<dim3(256 / 128, NPIX / 128), 256, 0, stream>>>(
        tokb, wot, bo, resf, nullptr, (int)NPIX, 256, 256);

    // --- y = LN(xe + res)  (also bf16 copy into xeb) ---
    ln_kernel<true><<<(unsigned)NPIX, 256, 0, stream>>>(xef, resf, g1, be1, yf, xeb);

    // --- h = relu(y @ w1 + b1)  (N=1024, K=256) -> bf16 into catb region ---
    gemm_bf16_kernel<true, false, true><<<dim3(1024 / 128, NPIX / 128), 256, 0, stream>>>(
        xeb, w1t, b1, nullptr, catb, (int)NPIX, 1024, 256);

    // --- r2 = h @ w2 + b2  (N=256, K=1024) -> f32 ---
    gemm_bf16_kernel<false, true, false><<<dim3(256 / 128, NPIX / 128), 256, 0, stream>>>(
        catb, w2t, b2, resf, nullptr, (int)NPIX, 256, 1024);

    // --- out = LN(y + r2) ---
    ln_kernel<false><<<(unsigned)NPIX, 256, 0, stream>>>(yf, resf, g2, be2, (float*)d_out, nullptr);
}
